// SwinTransformerBlock_15668040696332
// MI455X (gfx1250) — matrix operand order repaired and checked by reading, not compiled, not run on hardware
//
#include <hip/hip_runtime.h>
#include <math.h>

// ---------------------------------------------------------------------------
// CDNA5 (gfx1250) Swin block: bf16 WMMA GEMMs + fused attention.
// wave32 only; WMMA shape 16x16x32 (bf16 in, f32 acc).
// GEMM tiles staged via GLOBAL_LOAD_ASYNC_TO_LDS_B128 (ASYNCcnt path).
// ---------------------------------------------------------------------------

typedef __bf16 bf16;
typedef __attribute__((ext_vector_type(8)))  __bf16 v8bf;
typedef __attribute__((ext_vector_type(16))) __bf16 v16bf;
typedef __attribute__((ext_vector_type(8)))  float  v8f;

__device__ __forceinline__ v8f wmma_bf16(v16bf a, v16bf b, v8f c) {
  // (neg_a, A, neg_b, B, c_mod, C, reuse_a, reuse_b)
  return __builtin_amdgcn_wmma_f32_16x16x32_bf16(false, a, false, b, (short)0, c,
                                                 false, false);
}

__device__ __forceinline__ v8f v8f_zero() {
  v8f z;
#pragma unroll
  for (int i = 0; i < 8; ++i) z[i] = 0.0f;
  return z;
}

// Async copy of 16 bytes global -> LDS, per-lane addresses. Tracked by ASYNCcnt.
__device__ __forceinline__ void async_copy_b128(const void* gptr, void* lptr) {
  const unsigned long long ga = (unsigned long long)(size_t)gptr;
  const unsigned la = (unsigned)(size_t)lptr;  // low 32 bits = LDS offset
  asm volatile("global_load_async_to_lds_b128 %0, %1, off"
               :
               : "v"(la), "v"(ga)
               : "memory");
}

__device__ __forceinline__ void wait_async_zero() {
  asm volatile("s_wait_asynccnt 0" ::: "memory");
}

// ---------------------------------------------------------------------------
// fp32 -> bf16 conversion (weights)
// ---------------------------------------------------------------------------
__global__ __launch_bounds__(256) void cvt_bf16_kernel(const float* __restrict__ in,
                                                       bf16* __restrict__ out, int n) {
  int i = blockIdx.x * 256 + threadIdx.x;
  if (i < n) out[i] = (bf16)in[i];
}

// ---------------------------------------------------------------------------
// LayerNorm over C=384, 128 threads x 3 elements.
// GATHER=true: LN1 with roll(-3,-3) + window partition gather from x (BHWC).
// GATHER=false: LN2 plain row-major over xres.
// ---------------------------------------------------------------------------
template <bool GATHER>
__global__ __launch_bounds__(128) void ln384_kernel(const float* __restrict__ src,
                                                    const float* __restrict__ g,
                                                    const float* __restrict__ b,
                                                    bf16* __restrict__ dst) {
  __shared__ float s1[128], s2[128];
  const int tid = threadIdx.x;
  const int t = blockIdx.x;  // token id (windowed order if GATHER)

  const float* px;
  if (GATHER) {
    const int wnd = t / 49, n = t - wnd * 49;
    const int bimg = wnd >> 6, wi = wnd & 63;
    const int y = ((wi >> 3) * 7 + n / 7 + 3) % 56;   // roll(-3): gather +3
    const int xq = ((wi & 7) * 7 + n % 7 + 3) % 56;
    px = src + (((long)bimg * 56 + y) * 56 + xq) * 384;
  } else {
    px = src + (long)t * 384;
  }

  float v[3], s = 0.f, q = 0.f;
#pragma unroll
  for (int i = 0; i < 3; ++i) {
    v[i] = px[tid + i * 128];
    s += v[i];
    q += v[i] * v[i];
  }
  s1[tid] = s; s2[tid] = q;
  __syncthreads();
  for (int o = 64; o > 0; o >>= 1) {
    if (tid < o) { s1[tid] += s1[tid + o]; s2[tid] += s2[tid + o]; }
    __syncthreads();
  }
  const float mean = s1[0] * (1.f / 384.f);
  const float var = s2[0] * (1.f / 384.f) - mean * mean;
  const float rstd = rsqrtf(var + 1e-5f);
  bf16* out = dst + (long)t * 384;
#pragma unroll
  for (int i = 0; i < 3; ++i) {
    const int ci = tid + i * 128;
    out[ci] = (bf16)((v[i] - mean) * rstd * g[ci] + b[ci]);
  }
}

// LN over D=1536, in place on bf16 buffer. 256 threads x 6.
__global__ __launch_bounds__(256) void ln1536_kernel(bf16* __restrict__ m,
                                                     const float* __restrict__ g,
                                                     const float* __restrict__ b) {
  __shared__ float s1[256], s2[256];
  const int tid = threadIdx.x;
  bf16* row = m + (long)blockIdx.x * 1536;
  float v[6], s = 0.f, q = 0.f;
#pragma unroll
  for (int i = 0; i < 6; ++i) {
    v[i] = (float)row[tid + i * 256];
    s += v[i];
    q += v[i] * v[i];
  }
  s1[tid] = s; s2[tid] = q;
  __syncthreads();
  for (int o = 128; o > 0; o >>= 1) {
    if (tid < o) { s1[tid] += s1[tid + o]; s2[tid] += s2[tid + o]; }
    __syncthreads();
  }
  const float mean = s1[0] * (1.f / 1536.f);
  const float var = s2[0] * (1.f / 1536.f) - mean * mean;
  const float rstd = rsqrtf(var + 1e-5f);
#pragma unroll
  for (int i = 0; i < 6; ++i) {
    const int ci = tid + i * 256;
    row[ci] = (bf16)((v[i] - mean) * rstd * g[ci] + b[ci]);
  }
}

// ---------------------------------------------------------------------------
// Generic bf16 WMMA GEMM: C = A(MxK) * B(KxN) + bias, templated epilogue.
// Block tile 128x128, BK=32, 256 threads = 8 waves; each wave: 2x4 WMMA tiles.
// Tiles staged with global_load_async_to_lds_b128 (no VGPR round-trip).
// ---------------------------------------------------------------------------
enum { EPI_QKV = 0, EPI_PROJ = 1, EPI_FC1 = 2, EPI_FC2 = 3 };

template <int EPI>
__global__ __launch_bounds__(256) void gemm_wmma_kernel(
    const bf16* __restrict__ A, const bf16* __restrict__ B,
    const float* __restrict__ bias, int K, int N,
    bf16* __restrict__ out_bf, float* __restrict__ out_f,
    const float* __restrict__ aux) {
  __shared__ __align__(64) bf16 As[128][40];   // 128x32 tile, padded
  __shared__ __align__(64) bf16 Bs[32][144];   // 32x128 tile, padded (32B-aligned cols)

  const int tid = threadIdx.x;
  const int lane = tid & 31;
  const int wv = tid >> 5;               // wave 0..7
  const int m0 = (wv >> 1) * 32;         // wave row band in tile
  const int n0 = (wv & 1) * 64;          // wave col band in tile
  const long mBlk = (long)blockIdx.y * 128;
  const int nBlk = blockIdx.x * 128;

  v8f acc[2][4];
#pragma unroll
  for (int r = 0; r < 2; ++r)
#pragma unroll
    for (int c = 0; c < 4; ++c) acc[r][c] = v8f_zero();

  const int ksteps = K >> 5;
  for (int kt = 0; kt < ksteps; ++kt) {
    const int k0 = kt << 5;
    // --- async staged loads: global -> LDS directly (ASYNCcnt) ---
#pragma unroll
    for (int c = 0; c < 2; ++c) {
      const int ch = tid + c * 256;          // 512 chunks of 8 bf16
      const int row = ch >> 2, col = (ch & 3) << 3;
      async_copy_b128(A + (mBlk + row) * K + k0 + col, &As[row][col]);
    }
#pragma unroll
    for (int c = 0; c < 2; ++c) {
      const int ch = tid + c * 256;
      const int row = ch >> 4, col = (ch & 15) << 3;
      async_copy_b128(B + (long)(k0 + row) * N + nBlk + col, &Bs[row][col]);
    }
    wait_async_zero();   // own wave's async copies landed in LDS
    __syncthreads();     // all waves' copies visible

    if (kt + 1 < ksteps) {  // gfx1250 global_prefetch_b8 for next K tile
      __builtin_prefetch(A + (mBlk + (tid >> 1)) * K + k0 + 32, 0, 0);
      __builtin_prefetch(B + (long)(k0 + 32 + (tid >> 4)) * N + nBlk + ((tid & 15) << 3), 0, 0);
    }

    // --- fragments per ISA VGPR layout ---
    v16bf af[2], bfv[4];
    const int mA = lane & 15;
    const int kb = (lane & 16) ? 8 : 0;  // lanes 0-15: K 0-7/16-23; 16-31: 8-15/24-31
#pragma unroll
    for (int r = 0; r < 2; ++r) {
      v8bf lo = *(const v8bf*)&As[m0 + r * 16 + mA][kb];
      v8bf hi = *(const v8bf*)&As[m0 + r * 16 + mA][kb + 16];
#pragma unroll
      for (int i = 0; i < 8; ++i) { af[r][i] = lo[i]; af[r][i + 8] = hi[i]; }
    }
#pragma unroll
    for (int c = 0; c < 4; ++c)
#pragma unroll
      for (int i = 0; i < 8; ++i) {  // lane = N col; K in table order
        bfv[c][i] = Bs[kb + i][n0 + c * 16 + mA];
        bfv[c][i + 8] = Bs[kb + 16 + i][n0 + c * 16 + mA];
      }

#pragma unroll
    for (int r = 0; r < 2; ++r)
#pragma unroll
      for (int c = 0; c < 4; ++c) acc[r][c] = wmma_bf16(af[r], bfv[c], acc[r][c]);
    __syncthreads();
  }

  // --- epilogue: D layout lane->(n=lane&15, m=v + 8*(lane>>4)) ---
  const int nlo = lane & 15;
  const int mhi = (lane >> 4) << 3;
#pragma unroll
  for (int r = 0; r < 2; ++r)
#pragma unroll
    for (int c = 0; c < 4; ++c)
#pragma unroll
      for (int v = 0; v < 8; ++v) {
        const long row = mBlk + m0 + r * 16 + mhi + v;
        const int col = nBlk + n0 + c * 16 + nlo;
        float val = acc[r][c][v] + bias[col];
        if constexpr (EPI == EPI_QKV) {
          out_bf[row * N + col] = (bf16)val;
        } else if constexpr (EPI == EPI_PROJ) {
          // row = windowed token; scatter with window-reverse + roll(+3) + residual
          const int t = (int)row;
          const int wnd = t / 49, n = t - wnd * 49;
          const int bimg = wnd >> 6, wi = wnd & 63;
          const int y = ((wi >> 3) * 7 + n / 7 + 3) % 56;
          const int xq = ((wi & 7) * 7 + n % 7 + 3) % 56;
          const long img = (((long)bimg * 56 + y) * 56 + xq) * 384 + col;
          out_f[img] = val + aux[img];
        } else if constexpr (EPI == EPI_FC1) {
          const float ge = 0.5f * val * (1.0f + erff(val * 0.70710678118654752f));
          out_bf[row * N + col] = (bf16)ge;
        } else {  // EPI_FC2: + residual, final output
          const long idx = row * N + col;
          out_f[idx] = val + aux[idx];
        }
      }
}

// ---------------------------------------------------------------------------
// Window attention: one block (2 waves, 64 threads) per (window, head).
// N=49 padded to 64; hd=96. scores: 24 WMMAs; P@V: 24 WMMAs.
// Fused rel-pos bias + shift mask + softmax in LDS.
// ---------------------------------------------------------------------------
struct __align__(64) AttnSmem {
  union U {
    struct QK { bf16 Qs[64][104]; bf16 KTs[96][80]; } qk;  // 28672 B
    float Ps[64][65];                                      // 16640 B (reuse)
  } u;
  bf16 Vs[64][112];   // 14336 B
  bf16 Pbf[64][80];   // 10240 B
};

__global__ __launch_bounds__(64) void attn_kernel(const bf16* __restrict__ qkv,
                                                  const float* __restrict__ rpb,
                                                  bf16* __restrict__ o_win) {
  __shared__ AttnSmem sm;
  const int w = blockIdx.x >> 2;   // window 0..2047
  const int hh = blockIdx.x & 3;   // head
  const int tid = threadIdx.x;
  const int lane = tid & 31;
  const int wv = tid >> 5;         // wave 0/1 -> rows 0-31 / 32-63
  const float scale = 0.10206207262f;  // 96^-0.5

  // ---- stage Q (scaled), K^T, V into LDS; pad rows 49..63 with zeros ----
  for (int i = tid; i < 64 * 96; i += 64) {
    const int n = i / 96, d = i - n * 96;
    float q = 0.f, k = 0.f, v = 0.f;
    if (n < 49) {
      const long base = ((long)w * 49 + n) * 1152 + hh * 96 + d;
      q = (float)qkv[base] * scale;
      k = (float)qkv[base + 384];
      v = (float)qkv[base + 768];
    }
    sm.u.qk.Qs[n][d] = (bf16)q;
    sm.u.qk.KTs[d][n] = (bf16)k;
    sm.Vs[n][d] = (bf16)v;
  }
  __syncthreads();

  // ---- scores = Q @ K^T : 64x64, K-dim 96 (3 chunks) ----
  const int m0 = wv * 32;
  v8f acc[2][4];
#pragma unroll
  for (int r = 0; r < 2; ++r)
#pragma unroll
    for (int c = 0; c < 4; ++c) acc[r][c] = v8f_zero();

  const int mA = lane & 15;
  const int kboff = (lane & 16) ? 8 : 0;
#pragma unroll
  for (int kc = 0; kc < 3; ++kc) {
    v16bf af[2], bfv[4];
#pragma unroll
    for (int r = 0; r < 2; ++r) {
      v8bf lo = *(const v8bf*)&sm.u.qk.Qs[m0 + r * 16 + mA][kc * 32 + kboff];
      v8bf hi = *(const v8bf*)&sm.u.qk.Qs[m0 + r * 16 + mA][kc * 32 + kboff + 16];
#pragma unroll
      for (int i = 0; i < 8; ++i) { af[r][i] = lo[i]; af[r][i + 8] = hi[i]; }
    }
#pragma unroll
    for (int c = 0; c < 4; ++c)
#pragma unroll
      for (int i = 0; i < 8; ++i) {
        bfv[c][i] = sm.u.qk.KTs[kc * 32 + kboff + i][c * 16 + mA];
        bfv[c][i + 8] = sm.u.qk.KTs[kc * 32 + kboff + 16 + i][c * 16 + mA];
      }
#pragma unroll
    for (int r = 0; r < 2; ++r)
#pragma unroll
      for (int c = 0; c < 4; ++c) acc[r][c] = wmma_bf16(af[r], bfv[c], acc[r][c]);
  }
  __syncthreads();  // Qs/KTs dead -> safe to overlay Ps

  const int nlo = lane & 15;
  const int mhi = (lane >> 4) << 3;
#pragma unroll
  for (int r = 0; r < 2; ++r)
#pragma unroll
    for (int c = 0; c < 4; ++c)
#pragma unroll
      for (int v = 0; v < 8; ++v)
        sm.u.Ps[m0 + r * 16 + mhi + v][c * 16 + nlo] = acc[r][c][v];
  __syncthreads();

  // ---- softmax row per thread with rel-pos bias + shift mask ----
  {
    const int i = tid;
    if (i < 49) {
      const int wi = w & 63;
      const int bh = wi >> 3, bw = wi & 7;
      const int ih = i / 7, iw = i - ih * 7;
      const int yi = bh * 7 + ih, xi = bw * 7 + iw;
      const int ri = 3 * (yi < 49 ? 0 : (yi < 53 ? 1 : 2)) + (xi < 49 ? 0 : (xi < 53 ? 1 : 2));
      float rowv[64];
      float mx = -1e30f;
#pragma unroll
      for (int j = 0; j < 64; ++j) {
        float s;
        if (j < 49) {
          const int jh = j / 7, jw = j - jh * 7;
          const int yj = bh * 7 + jh, xj = bw * 7 + jw;
          const int rj = 3 * (yj < 49 ? 0 : (yj < 53 ? 1 : 2)) + (xj < 49 ? 0 : (xj < 53 ? 1 : 2));
          const int idx = (ih - jh + 6) * 13 + (iw - jw + 6);
          s = sm.u.Ps[i][j] + rpb[idx * 4 + hh] + ((ri == rj) ? 0.f : -100.f);
        } else {
          s = -1e30f;
        }
        rowv[j] = s;
        mx = fmaxf(mx, s);
      }
      float sum = 0.f;
#pragma unroll
      for (int j = 0; j < 64; ++j) { rowv[j] = __expf(rowv[j] - mx); sum += rowv[j]; }
      const float inv = 1.f / sum;
#pragma unroll
      for (int j = 0; j < 64; ++j) sm.Pbf[i][j] = (bf16)(rowv[j] * inv);
    } else {
#pragma unroll
      for (int j = 0; j < 64; ++j) sm.Pbf[i][j] = (bf16)0.f;
    }
  }
  __syncthreads();

  // ---- O = P @ V : 64x96, K-dim 64 (2 chunks) ----
  v8f acc2[2][6];
#pragma unroll
  for (int r = 0; r < 2; ++r)
#pragma unroll
    for (int c = 0; c < 6; ++c) acc2[r][c] = v8f_zero();
#pragma unroll
  for (int kc = 0; kc < 2; ++kc) {
    v16bf af[2], bfv[6];
#pragma unroll
    for (int r = 0; r < 2; ++r) {
      v8bf lo = *(const v8bf*)&sm.Pbf[m0 + r * 16 + mA][kc * 32 + kboff];
      v8bf hi = *(const v8bf*)&sm.Pbf[m0 + r * 16 + mA][kc * 32 + kboff + 16];
#pragma unroll
      for (int i = 0; i < 8; ++i) { af[r][i] = lo[i]; af[r][i + 8] = hi[i]; }
    }
#pragma unroll
    for (int c = 0; c < 6; ++c)
#pragma unroll
      for (int i = 0; i < 8; ++i) {
        bfv[c][i] = sm.Vs[kc * 32 + kboff + i][c * 16 + mA];
        bfv[c][i + 8] = sm.Vs[kc * 32 + kboff + 16 + i][c * 16 + mA];
      }
#pragma unroll
    for (int r = 0; r < 2; ++r)
#pragma unroll
      for (int c = 0; c < 6; ++c) acc2[r][c] = wmma_bf16(af[r], bfv[c], acc2[r][c]);
  }

  // ---- write O rows < 49 ----
#pragma unroll
  for (int r = 0; r < 2; ++r)
#pragma unroll
    for (int c = 0; c < 6; ++c)
#pragma unroll
      for (int v = 0; v < 8; ++v) {
        const int n = m0 + r * 16 + mhi + v;
        if (n < 49) {
          const int d = c * 16 + nlo;
          o_win[((long)w * 49 + n) * 384 + hh * 96 + d] = (bf16)acc2[r][c][v];
        }
      }
}

// ---------------------------------------------------------------------------
// Host-side orchestration
// ---------------------------------------------------------------------------
extern "C" void kernel_launch(void* const* d_in, const int* in_sizes, int n_in,
                              void* d_out, int out_size, void* d_ws, size_t ws_size,
                              hipStream_t stream) {
  const float* x      = (const float*)d_in[0];
  const float* ln1_g  = (const float*)d_in[1];
  const float* ln1_b  = (const float*)d_in[2];
  const float* qkv_w  = (const float*)d_in[3];
  const float* qkv_b  = (const float*)d_in[4];
  const float* proj_w = (const float*)d_in[5];
  const float* proj_b = (const float*)d_in[6];
  const float* rpb    = (const float*)d_in[7];
  const float* ln2_g  = (const float*)d_in[8];
  const float* ln2_b  = (const float*)d_in[9];
  const float* fc1_w  = (const float*)d_in[10];
  const float* fc1_b  = (const float*)d_in[11];
  const float* ln3_g  = (const float*)d_in[12];
  const float* ln3_b  = (const float*)d_in[13];
  const float* fc2_w  = (const float*)d_in[14];
  const float* fc2_b  = (const float*)d_in[15];

  const long T = 100352;  // 32*56*56 tokens

  char* ws = (char*)d_ws;
  size_t off = 0;
  auto take = [&](size_t bytes) -> void* {
    void* p = ws + off;
    off += (bytes + 255) & ~(size_t)255;
    return p;
  };
  bf16*  w_qkv  = (bf16*)take(442368ull * 2);
  bf16*  w_proj = (bf16*)take(147456ull * 2);
  bf16*  w_fc1  = (bf16*)take(589824ull * 2);
  bf16*  w_fc2  = (bf16*)take(589824ull * 2);
  bf16*  h_win  = (bf16*)take((size_t)T * 384 * 2);
  bf16*  qkvb   = (bf16*)take((size_t)T * 1152 * 2);
  bf16*  o_win  = (bf16*)take((size_t)T * 384 * 2);
  float* xres   = (float*)take((size_t)T * 384 * 4);
  bf16*  h2     = (bf16*)take((size_t)T * 384 * 2);
  bf16*  m1     = (bf16*)take((size_t)T * 1536 * 2);

  // 0) weights to bf16
  cvt_bf16_kernel<<<(442368 + 255) / 256, 256, 0, stream>>>(qkv_w, w_qkv, 442368);
  cvt_bf16_kernel<<<(147456 + 255) / 256, 256, 0, stream>>>(proj_w, w_proj, 147456);
  cvt_bf16_kernel<<<(589824 + 255) / 256, 256, 0, stream>>>(fc1_w, w_fc1, 589824);
  cvt_bf16_kernel<<<(589824 + 255) / 256, 256, 0, stream>>>(fc2_w, w_fc2, 589824);

  // 1) LN1 + roll(-3) + window partition -> bf16
  ln384_kernel<true><<<(int)T, 128, 0, stream>>>(x, ln1_g, ln1_b, h_win);

  // 2) qkv = h @ qkv_w + b   (100352x384 @ 384x1152)
  gemm_wmma_kernel<EPI_QKV><<<dim3(1152 / 128, (int)(T / 128)), 256, 0, stream>>>(
      h_win, w_qkv, qkv_b, 384, 1152, qkvb, nullptr, nullptr);

  // 3) windowed shifted attention (2048 windows x 4 heads)
  attn_kernel<<<2048 * 4, 64, 0, stream>>>(qkvb, rpb, o_win);

  // 4) proj + window-reverse + roll(+3) + residual -> xres (f32, image layout)
  gemm_wmma_kernel<EPI_PROJ><<<dim3(384 / 128, (int)(T / 128)), 256, 0, stream>>>(
      o_win, w_proj, proj_b, 384, 384, nullptr, xres, x);

  // 5) LN2 -> bf16
  ln384_kernel<false><<<(int)T, 128, 0, stream>>>(xres, ln2_g, ln2_b, h2);

  // 6) fc1 + exact GELU -> bf16 (100352x384 @ 384x1536)
  gemm_wmma_kernel<EPI_FC1><<<dim3(1536 / 128, (int)(T / 128)), 256, 0, stream>>>(
      h2, w_fc1, fc1_b, 384, 1536, m1, nullptr, nullptr);

  // 7) LN3 over D=1536, in place
  ln1536_kernel<<<(int)T, 256, 0, stream>>>(m1, ln3_g, ln3_b);

  // 8) fc2 + bias + residual -> d_out (100352x1536 @ 1536x384)
  gemm_wmma_kernel<EPI_FC2><<<dim3(384 / 128, (int)(T / 128)), 256, 0, stream>>>(
      m1, w_fc2, fc2_b, 1536, 384, nullptr, (float*)d_out, xres);

  (void)in_sizes; (void)n_in; (void)out_size; (void)ws_size;
}
